// NISRec_4312147165791
// MI455X (gfx1250) — compile-verified, matrix operand order checked by reading, not run on hardware
//
#include <hip/hip_runtime.h>

// ---------------------------------------------------------------------------
// NISRec fused attention for MI455X (gfx1250, wave32, WMMA 16x16x32 f16)
//
//   Q    = queries @ Qw^T + Qb                       -> f16 [B,L,H]
//   Ks   = keys    @ Kw^T + Kb + nbr + nat + pk      -> f16 [B,L,H]
//   VsT  = (keys   @ Vw^T + Vb + nbr + nat + pv)^T   -> f16 [B,NH,HD,L]
//   out  = softmax(causal(Q Ks^T / 8)) @ Vs          -> f32 [B,L,H]
//
// Flash-attention computes S^T = K_tile @ Q^T so that P^T is already in the
// WMMA B-fragment layout for out^T = V^T @ P^T (pure in-lane cvt, no LDS).
// ---------------------------------------------------------------------------

typedef __attribute__((ext_vector_type(16))) _Float16 v16h;
typedef __attribute__((ext_vector_type(8)))  _Float16 v8h;
typedef __attribute__((ext_vector_type(8)))  float    v8f;

#define BB  128
#define LL  512
#define HH  128
#define NHH 2
#define HDD 64
#define BL  (BB * LL)

__device__ __forceinline__ v8f wmma_f16(v16h a, v16h b, v8f c) {
  return __builtin_amdgcn_wmma_f32_16x16x32_f16(
      /*neg_a=*/false, a, /*neg_b=*/false, b,
      /*c_mod=*/(short)0, c, /*reuse_a=*/false, /*reuse_b=*/false);
}

// A/B fragment loader from f16 row-major data.
// p = row_base + k_chunk + (lane>>4)*8 ; elements 0..7 <- p[0..7],
// elements 8..15 <- p[16..23]  (matches 16-bit 16x32 fragment layout).
__device__ __forceinline__ v16h frag_h(const _Float16* p) {
  v8h lo = *(const v8h*)(p);
  v8h hi = *(const v8h*)(p + 16);
  return __builtin_shufflevector(lo, hi,
      0, 1, 2, 3, 4, 5, 6, 7, 8, 9, 10, 11, 12, 13, 14, 15);
}

// Same fragment, sourced from f32 data with on-the-fly f32->f16 conversion.
__device__ __forceinline__ v16h frag_f32(const float* p) {
  float4 x0 = *(const float4*)(p);
  float4 x1 = *(const float4*)(p + 4);
  float4 x2 = *(const float4*)(p + 16);
  float4 x3 = *(const float4*)(p + 20);
  v16h f;
  f[0]  = (_Float16)x0.x; f[1]  = (_Float16)x0.y;
  f[2]  = (_Float16)x0.z; f[3]  = (_Float16)x0.w;
  f[4]  = (_Float16)x1.x; f[5]  = (_Float16)x1.y;
  f[6]  = (_Float16)x1.z; f[7]  = (_Float16)x1.w;
  f[8]  = (_Float16)x2.x; f[9]  = (_Float16)x2.y;
  f[10] = (_Float16)x2.z; f[11] = (_Float16)x2.w;
  f[12] = (_Float16)x3.x; f[13] = (_Float16)x3.y;
  f[14] = (_Float16)x3.z; f[15] = (_Float16)x3.w;
  return f;
}

// ---------------------------------------------------------------------------
// Projection: out16 = f16( X @ W^T + bias [+ a1 + a2 + a3] )
// One wave computes one 16x16 output tile with 4 x v_wmma_f32_16x16x32_f16.
// transposed=1 stores [B,NH,HD,L] (for the V^T operand of attention).
// ---------------------------------------------------------------------------
__global__ void __launch_bounds__(256)
nisrec_proj_kernel(const float* __restrict__ X, const float* __restrict__ W,
                   const float* __restrict__ bias,
                   const float* __restrict__ a1, const float* __restrict__ a2,
                   const float* __restrict__ a3,
                   _Float16* __restrict__ out, int transposed)
{
  const int lane = threadIdx.x & 31;
  const int wave = threadIdx.x >> 5;
  const int l16  = lane & 15;
  const int half = lane >> 4;

  const int tid = blockIdx.x * 8 + wave;     // 32768 tiles total
  const int mt  = tid >> 3;                  // 4096 M tiles (BL/16)
  const int nt  = tid & 7;                   // 8    N tiles (H/16)
  const int m0  = mt * 16, n0 = nt * 16;

  const float* arow = X + (size_t)(m0 + l16) * HH;   // A: X rows
  const float* brow = W + (size_t)(n0 + l16) * HH;   // B: W rows (= cols of W^T)

  v8f c = {};
#pragma unroll
  for (int kk = 0; kk < HH; kk += 32) {
    v16h a = frag_f32(arow + kk + half * 8);
    v16h b = frag_f32(brow + kk + half * 8);
    c = wmma_f16(a, b, c);
  }

  const int   cc = n0 + l16;
  const float bc = bias[cc];
#pragma unroll
  for (int g = 0; g < 8; ++g) {
    const int r = m0 + g + 8 * half;         // C layout: row = g + 8*(lane>>4)
    float v = c[g] + bc;
    if (a1) {
      const size_t idx = (size_t)r * HH + cc;
      v += a1[idx] + a2[idx] + a3[idx];
    }
    if (!transposed) {
      out[(size_t)r * HH + cc] = (_Float16)v;
    } else {
      const int b_ = r >> 9, l_ = r & (LL - 1);
      const int h_ = cc >> 6, d_ = cc & (HDD - 1);
      out[((size_t)(b_ * NHH + h_) * HDD + d_) * LL + l_] = (_Float16)v;
    }
  }
}

// ---------------------------------------------------------------------------
// Flash attention, causal, one wave per 16-query tile of one (batch, head).
//   S^T tile  : st = Ks_tile(16k x 64d) @ Q^T(64d x 16q)   (2 WMMA per 16 keys)
//   out^T acc : acc[t] += Vt_tile(16d x 32k) @ P^T(32k x 16q)
// P^T comes straight out of the S^T C-layout: in-lane f32->f16 only.
// ---------------------------------------------------------------------------
__global__ void __launch_bounds__(128)
nisrec_attn_kernel(const _Float16* __restrict__ Q, const _Float16* __restrict__ Ks,
                   const _Float16* __restrict__ VT, float* __restrict__ out)
{
  const int lane = threadIdx.x & 31;
  const int wave = threadIdx.x >> 5;
  const int l16  = lane & 15;
  const int half = lane >> 4;

  const int b   = blockIdx.x / (NHH * 8);
  const int rem = blockIdx.x % (NHH * 8);
  const int hh  = rem >> 3;
  const int q0  = ((rem & 7) * 4 + wave) * 16;   // 32 query tiles per (b,h)

  const _Float16* qbase = Q  + ((size_t)b * LL) * HH + hh * HDD;
  const _Float16* kbase = Ks + ((size_t)b * LL) * HH + hh * HDD;
  const _Float16* vtb   = VT + ((size_t)(b * NHH + hh) * HDD) * LL;

  // Q^T B-fragments: column q = q0 + l16, d split 0..31 / 32..63
  const _Float16* qrow = qbase + (size_t)(q0 + l16) * HH + half * 8;
  const v16h bq0 = frag_h(qrow);
  const v16h bq1 = frag_h(qrow + 32);

  v8f acc0 = {}, acc1 = {}, acc2 = {}, acc3 = {};
  const float NEGINF = -__builtin_inff();
  float m = NEGINF, lsum = 0.0f;
  const int q      = q0 + l16;
  const int nsteps = (q0 + 47) >> 5;           // keys 0 .. q0+15, 32 per step

  for (int kt = 0; kt < nsteps; ++kt) {
    const int k0 = kt * 32;
    const _Float16* krow0 = kbase + (size_t)(k0 + l16) * HH + half * 8;
    const _Float16* krow1 = krow0 + (size_t)16 * HH;
    if (kt + 1 < nsteps) {                      // hint next key/value tiles
      __builtin_prefetch(krow0 + (size_t)32 * HH, 0, 3);
    }

    // S^T = Ks_tile @ Q^T  (two 16-key subtiles, K-dim = HD = 64)
    v8f st0 = {}, st1 = {};
    st0 = wmma_f16(frag_h(krow0), bq0, st0);
    st0 = wmma_f16(frag_h(krow0 + 32), bq1, st0);
    st1 = wmma_f16(frag_h(krow1), bq0, st1);
    st1 = wmma_f16(frag_h(krow1 + 32), bq1, st1);

    // scale + causal mask; C layout: key row = k0 + g + 8*half (+16 for st1),
    // query col = l16 (shared by lane and lane^16).
    float s0[8], s1[8];
    float tmax = NEGINF;
#pragma unroll
    for (int g = 0; g < 8; ++g) {
      const int kr = k0 + g + 8 * half;
      float x0 = st0[g] * 0.125f;
      float x1 = st1[g] * 0.125f;
      if (kr > q)       x0 = NEGINF;
      if (kr + 16 > q)  x1 = NEGINF;
      s0[g] = x0; s1[g] = x1;
      tmax = fmaxf(tmax, fmaxf(x0, x1));
    }
    tmax = fmaxf(tmax, __shfl_xor(tmax, 16, 32));

    const float mnew  = fmaxf(m, tmax);
    const float alpha = __expf(m - mnew);       // first iter: exp(-inf) = 0
    m = mnew;

    float p0[8], p1[8], rs = 0.0f;
#pragma unroll
    for (int g = 0; g < 8; ++g) {
      p0[g] = __expf(s0[g] - mnew);
      p1[g] = __expf(s1[g] - mnew);
      rs += p0[g] + p1[g];
    }
    rs += __shfl_xor(rs, 16, 32);
    lsum = lsum * alpha + rs;

#pragma unroll
    for (int g = 0; g < 8; ++g) {
      acc0[g] *= alpha; acc1[g] *= alpha; acc2[g] *= alpha; acc3[g] *= alpha;
    }

    // P^T B-fragment: lane n holds keys {8h..8h+7} and {16+8h..16+8h+7}
    // for query n%16 — exactly the values sitting in st0/st1 on this lane.
    v16h bp;
#pragma unroll
    for (int g = 0; g < 8; ++g) {
      bp[g]     = (_Float16)p0[g];
      bp[8 + g] = (_Float16)p1[g];
    }

    // out^T += V^T(16d x 32k) @ P^T, four 16-row d tiles
    const _Float16* vrow = vtb + (size_t)l16 * LL + k0 + half * 8;
    if (kt + 1 < nsteps) __builtin_prefetch(vrow + 32, 0, 3);
    acc0 = wmma_f16(frag_h(vrow),                   bp, acc0);
    acc1 = wmma_f16(frag_h(vrow + (size_t)16 * LL), bp, acc1);
    acc2 = wmma_f16(frag_h(vrow + (size_t)32 * LL), bp, acc2);
    acc3 = wmma_f16(frag_h(vrow + (size_t)48 * LL), bp, acc3);
  }

  // acc C layout: row d = t*16 + g + 8*half, col q = l16 -> per-lane the 8
  // elements of each tile are contiguous d's of one output row: float4 stores.
  const float inv = 1.0f / lsum;
  float* orow = out + ((size_t)(b * LL) + q) * HH + hh * HDD;
  const int d0 = 8 * half;
#define NISREC_STORE(A, T)                                                     \
  {                                                                            \
    float4 u0 = make_float4(A[0] * inv, A[1] * inv, A[2] * inv, A[3] * inv);   \
    float4 u1 = make_float4(A[4] * inv, A[5] * inv, A[6] * inv, A[7] * inv);   \
    *(float4*)(orow + (T)*16 + d0)     = u0;                                   \
    *(float4*)(orow + (T)*16 + d0 + 4) = u1;                                   \
  }
  NISREC_STORE(acc0, 0)
  NISREC_STORE(acc1, 1)
  NISREC_STORE(acc2, 2)
  NISREC_STORE(acc3, 3)
#undef NISREC_STORE
}

// ---------------------------------------------------------------------------
extern "C" void kernel_launch(void* const* d_in, const int* in_sizes, int n_in,
                              void* d_out, int out_size, void* d_ws, size_t ws_size,
                              hipStream_t stream)
{
  (void)in_sizes; (void)n_in; (void)out_size; (void)ws_size;

  const float* queries = (const float*)d_in[0];
  const float* keys    = (const float*)d_in[1];
  const float* nbr     = (const float*)d_in[2];
  const float* nat     = (const float*)d_in[3];
  const float* pk      = (const float*)d_in[4];
  const float* pv      = (const float*)d_in[5];
  const float* Qw      = (const float*)d_in[6];
  const float* Qb      = (const float*)d_in[7];
  const float* Kw      = (const float*)d_in[8];
  const float* Kb      = (const float*)d_in[9];
  const float* Vw      = (const float*)d_in[10];
  const float* Vb      = (const float*)d_in[11];
  // d_in[12] = causal mask; regenerated analytically inside the kernel.

  _Float16* q16  = (_Float16*)d_ws;                       // [B,L,H]    16.8 MB
  _Float16* k16  = q16  + (size_t)BL * HH;                // [B,L,H]    16.8 MB
  _Float16* vt16 = k16  + (size_t)BL * HH;                // [B,NH,HD,L]16.8 MB
  float*    out  = (float*)d_out;

  const int projBlocks = (BL / 16) * (HH / 16) / 8;       // 4096 blocks x 8 waves

  nisrec_proj_kernel<<<projBlocks, 256, 0, stream>>>(
      queries, Qw, Qb, nullptr, nullptr, nullptr, q16, 0);
  nisrec_proj_kernel<<<projBlocks, 256, 0, stream>>>(
      keys, Kw, Kb, nbr, nat, pk, k16, 0);
  nisrec_proj_kernel<<<projBlocks, 256, 0, stream>>>(
      keys, Vw, Vb, nbr, nat, pv, vt16, 1);

  nisrec_attn_kernel<<<BB * NHH * 8, 128, 0, stream>>>(q16, k16, vt16, out);
}
